// ConvAttLIF_40973988004675
// MI455X (gfx1250) — matrix operand order repaired and checked by reading, not compiled, not run on hardware
//
#include <hip/hip_runtime.h>
#include <hip/hip_bf16.h>
#include <stdint.h>

// MI455X / gfx1250 (CDNA5, wave32). Conv = implicit GEMM on
// v_wmma_f32_16x16x32_bf16, staged via global_load_async_to_lds_b128.
// ~72.5 GFLOP conv, ~630 MB min HBM traffic -> balanced at 23.3 TB/s.

typedef __attribute__((ext_vector_type(8)))  __bf16 v8bf;
typedef __attribute__((ext_vector_type(16))) __bf16 v16bf;
typedef __attribute__((ext_vector_type(8)))  float  v8f;

#define B_    8
#define T_    60
#define CIN   64
#define COUT  128
#define KTOT  576      // 9 * CIN, K order: k = (kh*3+kw)*64 + ci
#define NCHUNK 18      // KTOT/32
#define APAD  40       // 32 K + 8 pad (bf16) -> 80B row stride, conflict-free b128 frags
#define TPAD  70       // 64 pix + 6 pad for transpose tile
#define NIMG  (B_ * T_)                       // 480
#define YELEMS  ((size_t)NIMG * 1024 * COUT)  // 62,914,560 bf16
#define XELEMS  ((size_t)NIMG * 1024 * CIN)   // 31,457,280 bf16
#define WELEMS  ((size_t)COUT * KTOT)         // 73,728 bf16

// LDS offsets (ushort units) inside one shared buffer:
#define AOFF(buf) ((buf) * 2560)            // A buffers: 64*APAD each
#define BOFF(buf) (5120 + (buf) * 5120)     // B buffers: 128*APAD each

__device__ __forceinline__ unsigned short f2bf(float x) {
  unsigned int u = __float_as_uint(x);
  u += 0x7FFFu + ((u >> 16) & 1u);   // RNE
  return (unsigned short)(u >> 16);
}
__device__ __forceinline__ float bf2f(unsigned short h) {
  return __uint_as_float(((unsigned int)h) << 16);
}
// Async 16B copy global -> LDS (CDNA5). ldsaddr = low 32 bits of generic
// shared pointer (ISA 10.2).
__device__ __forceinline__ void async_b128(const void* gsrc, const unsigned short* lds) {
  unsigned int laddr = (unsigned int)(uintptr_t)lds;
  asm volatile("global_load_async_to_lds_b128 %0, %1, off"
               :: "v"(laddr), "v"(gsrc) : "memory");
}

// ---------------------------------------------------------------------------
// Prep A: fp32 (n,ci,h,w) -> bf16 channels-last (n,h,w,ci). grid=480*32.
// ---------------------------------------------------------------------------
__global__ __launch_bounds__(256)
void xpose_kernel(const float* __restrict__ data, unsigned short* __restrict__ xbf) {
  __shared__ unsigned short t[64 * 33];
  int n = blockIdx.x >> 5, h = blockIdx.x & 31, tid = threadIdx.x;
  for (int e = tid; e < 2048; e += 256) {              // coalesced fp32 reads
    int ci = e >> 5, w = e & 31;
    t[ci * 33 + w] = f2bf(data[(((size_t)n * CIN + ci) * 32 + h) * 32 + w]);
  }
  __syncthreads();
  for (int e = tid; e < 2048; e += 256) {              // coalesced bf16 writes
    int w = e >> 6, ci = e & 63;
    xbf[(((size_t)n * 32 + h) * 32 + w) * CIN + ci] = t[ci * 33 + w];
  }
}

// ---------------------------------------------------------------------------
// Prep B: weights (cout,ci,kh,kw) fp32 -> bf16 (cout, khkw*64+ci). grid=288.
// ---------------------------------------------------------------------------
__global__ __launch_bounds__(256)
void wprep_kernel(const float* __restrict__ convw, unsigned short* __restrict__ wbf) {
  int idx = blockIdx.x * 256 + threadIdx.x;
  if (idx < (int)WELEMS) {
    int n = idx / KTOT, k = idx % KTOT;
    int khkw = k >> 6, ci = k & 63;
    int kh = khkw / 3, kw = khkw - kh * 3;
    wbf[idx] = f2bf(convw[((n * CIN + ci) * 3 + kh) * 3 + kw]);
  }
}

// ---------------------------------------------------------------------------
// Kernel 1: implicit-GEMM conv. grid = 480 imgs * 16 m-tiles. 256 thr (8 waves).
// Per block: 64 output pixels x 128 Cout. A/B staged by async-LDS engine.
// ---------------------------------------------------------------------------
__global__ __launch_bounds__(256)
void conv_wmma_kernel(const unsigned short* __restrict__ xbf,
                      const unsigned short* __restrict__ wbf,
                      const float* __restrict__ convb,
                      const unsigned short* __restrict__ zerobuf,
                      unsigned short* __restrict__ yws,
                      float* __restrict__ psum,
                      float* __restrict__ pmax) {
  __shared__ unsigned short smem[15360];   // A0|A1|B0|B1, reused as transpose tile
  __shared__ float wred[16];

  const int tid  = threadIdx.x;
  const int lane = tid & 31;
  const int wave = tid >> 5;
  const int n_img   = blockIdx.x >> 4;
  const int mtile   = blockIdx.x & 15;
  const int pixbase = mtile * 64;
  const int mt    = wave >> 1;
  const int nhalf = wave & 1;

  v8f acc[4];
  {
    v8f vz = {0.f,0.f,0.f,0.f,0.f,0.f,0.f,0.f};
    acc[0] = vz; acc[1] = vz; acc[2] = vz; acc[3] = vz;
  }

  // Stage one 32-K chunk: exactly 3 async b128 per wave (fixed count so a
  // static s_wait_asynccnt 0x3 retires the *previous* stage).
  auto stage = [&](int buf, int kc) {
    const int ab = AOFF(buf), bb = BOFF(buf);
    const int khkw = kc >> 1;
    const int kh = khkw / 3, kw = khkw - kh * 3;
    const int cb = (kc & 1) * 32;
    {   // A: 64 pix x 32 ci, 8 ci per thread -> 1 async/thread
      int m = tid >> 2, j = tid & 3;
      int pix = pixbase + m;
      int ih = (pix >> 5) + kh - 1;
      int iw = (pix & 31) + kw - 1;
      bool inb = ((unsigned)ih < 32u) & ((unsigned)iw < 32u);
      const unsigned short* src = inb
          ? &xbf[(((size_t)n_img * 32 + ih) * 32 + iw) * CIN + cb + j * 8]
          : zerobuf;                                   // halo -> zeros
      async_b128(src, &smem[ab + m * APAD + j * 8]);
    }
#pragma unroll
    for (int it = 0; it < 2; ++it) {  // B: 128 cout x 32 K -> 2 async/thread
      int e = it * 256 + tid;
      int nn = e >> 2, j = e & 3;
      async_b128(&wbf[(size_t)nn * KTOT + kc * 32 + j * 8],
                 &smem[bb + nn * APAD + j * 8]);
    }
  };

  stage(0, 0);
  for (int kc = 0; kc < NCHUNK; ++kc) {
    if (kc + 1 < NCHUNK) {
      __syncthreads();                     // compute(kc-1) done -> safe to overwrite
      stage((kc + 1) & 1, kc + 1);         // prefetch next chunk (in flight)
      asm volatile("s_wait_asynccnt 0x3" ::: "memory");   // stage(kc) landed
    } else {
      asm volatile("s_wait_asynccnt 0x0" ::: "memory");
    }
    __syncthreads();                       // all waves' stage(kc) visible
    const int ab = AOFF(kc & 1), bb = BOFF(kc & 1);

    // A fragment (ISA 16-bit A 16x32 layout): lanes 0-15: K0-7|K16-23;
    // lanes 16-31: K8-15|K24-31; row M = lane%16.
    int am  = mt * 16 + (lane & 15);
    int koA = (lane < 16) ? 0 : 8;
    v8bf alo = *(const v8bf*)&smem[ab + am * APAD + koA];
    v8bf ahi = *(const v8bf*)&smem[ab + am * APAD + 16 + koA];
    v16bf afrag = __builtin_shufflevector(alo, ahi,
        0,1,2,3,4,5,6,7,8,9,10,11,12,13,14,15);

    int koB = (lane < 16) ? 0 : 8;        // B: col N = lane%16, same K layout as A
#pragma unroll
    for (int nb = 0; nb < 4; ++nb) {
      int bn = nhalf * 64 + nb * 16 + (lane & 15);
      v8bf blo = *(const v8bf*)&smem[bb + bn * APAD + koB];
      v8bf bhi = *(const v8bf*)&smem[bb + bn * APAD + koB + 16];
      v16bf bfrag = __builtin_shufflevector(blo, bhi,
          0,1,2,3,4,5,6,7,8,9,10,11,12,13,14,15);
      acc[nb] = __builtin_amdgcn_wmma_f32_16x16x32_bf16(
          false, afrag, false, bfrag, (short)0, acc[nb], false, false);
    }
  }
  __syncthreads();   // reuse smem as [COUT][TPAD] transpose tile

  float ps = 0.f, pm = -3.4e38f;
#pragma unroll
  for (int nb = 0; nb < 4; ++nb) {
    int n = nhalf * 64 + nb * 16 + (lane & 15);
    float bias = convb[n];
    union { v8f v; float f[8]; } u; u.v = acc[nb];
#pragma unroll
    for (int r = 0; r < 8; ++r) {
      // C/D: VGPR r, lanes 0-15 -> M=r, lanes 16-31 -> M=r+8; N = lane%16
      float val = u.f[r] + bias;
      int pl = mt * 16 + r + ((lane >> 4) << 3);
      smem[n * TPAD + pl] = f2bf(val);
      ps += val;
      pm = fmaxf(pm, val);
    }
  }
  for (int off = 16; off > 0; off >>= 1) {   // deterministic partials, no atomics
    ps += __shfl_xor(ps, off, 32);
    pm  = fmaxf(pm, __shfl_xor(pm, off, 32));
  }
  if (lane == 0) { wred[wave] = ps; wred[8 + wave] = pm; }
  __syncthreads();
  if (tid == 0) {
    float s = 0.f, mx = -3.4e38f;
    for (int i = 0; i < 8; ++i) { s += wred[i]; mx = fmaxf(mx, wred[8 + i]); }
    psum[n_img * 16 + mtile] = s;
    pmax[n_img * 16 + mtile] = mx;
  }
  for (int e = tid; e < 64 * 128; e += 256) {   // y: (img,pix,c), c contiguous
    int c = e & 127, pl = e >> 7;
    yws[(size_t)n_img * (1024 * COUT) + (size_t)(pixbase + pl) * COUT + c] =
        smem[c * TPAD + pl];
  }
}

// ---------------------------------------------------------------------------
// Kernel 2: temporal attention. grid = 8 (one block per batch), 64 threads.
// ---------------------------------------------------------------------------
__global__ void ta_att_kernel(const float* __restrict__ psum,
                              const float* __restrict__ pmax,
                              const float* __restrict__ w1,
                              const float* __restrict__ w2,
                              float* __restrict__ att) {
  __shared__ float pa[T_], pmx[T_], ha[3], hm[3];
  int b = blockIdx.x, t = threadIdx.x;
  if (t < T_) {
    float s = 0.f, mx = -3.4e38f;
    for (int i = 0; i < 16; ++i) {
      s  += psum[(b * T_ + t) * 16 + i];
      mx  = fmaxf(mx, pmax[(b * T_ + t) * 16 + i]);
    }
    pa[t]  = s * (1.f / (128.f * 1024.f));
    pmx[t] = mx;
  }
  __syncthreads();
  if (t < 3) {
    float sa = 0.f, sm = 0.f;
    for (int i = 0; i < T_; ++i) {
      sa += w1[t * T_ + i] * pa[i];
      sm += w1[t * T_ + i] * pmx[i];
    }
    ha[t] = fmaxf(sa, 0.f);
    hm[t] = fmaxf(sm, 0.f);
  }
  __syncthreads();
  if (t < T_) {
    float sa = 0.f, sm = 0.f;
    for (int j = 0; j < 3; ++j) {
      sa += w2[t * 3 + j] * ha[j];
      sm += w2[t * 3 + j] * hm[j];
    }
    att[b * T_ + t] = 1.f / (1.f + expf(-(sa + sm)));
  }
}

// ---------------------------------------------------------------------------
// Kernel 3: IF-neuron scan over T. grid = 8*64, 256 threads.
// ---------------------------------------------------------------------------
__global__ __launch_bounds__(256)
void lif_kernel(const unsigned short* __restrict__ yws,
                const float* __restrict__ att,
                float* __restrict__ out) {
  __shared__ float tile[16 * 129];
  int b  = blockIdx.x >> 6;
  int pb = blockIdx.x & 63;
  int tid = threadIdx.x;
  float v[8] = {0.f,0.f,0.f,0.f,0.f,0.f,0.f,0.f};
  for (int t = 0; t < T_; ++t) {
    float a = att[b * T_ + t];
    __syncthreads();
    size_t ybase = ((size_t)(b * T_ + t) * 1024 + (size_t)pb * 16) * COUT;
#pragma unroll
    for (int i = 0; i < 8; ++i) {
      int e = i * 256 + tid;
      int pl = e >> 7, c = e & 127;
      tile[pl * 129 + c] = bf2f(yws[ybase + (size_t)pl * COUT + c]) * a;
    }
    __syncthreads();
    size_t obase = (size_t)(b * T_ + t) * (COUT * 1024) + (size_t)pb * 16;
#pragma unroll
    for (int i = 0; i < 8; ++i) {
      int e = i * 256 + tid;
      int c = e >> 4, pl = e & 15;
      v[i] += tile[pl * 129 + c];
      float s = (v[i] >= 0.6f) ? 1.f : 0.f;
      out[obase + (size_t)c * 1024 + pl] = s;
      v[i] = (s > 0.f) ? 0.f : v[i];
    }
  }
}

// ---------------------------------------------------------------------------
extern "C" void kernel_launch(void* const* d_in, const int* in_sizes, int n_in,
                              void* d_out, int out_size, void* d_ws, size_t ws_size,
                              hipStream_t stream) {
  const float* data   = (const float*)d_in[0];
  const float* conv_w = (const float*)d_in[1];
  const float* conv_b = (const float*)d_in[2];
  const float* ta_w1  = (const float*)d_in[3];
  const float* ta_w2  = (const float*)d_in[4];
  float* out = (float*)d_out;

  char* ws = (char*)d_ws;
  unsigned short* yws = (unsigned short*)ws;                       // 125.8 MB
  unsigned short* xbf = (unsigned short*)(ws + YELEMS * 2);        //  62.9 MB
  unsigned short* wbf = (unsigned short*)(ws + YELEMS * 2 + XELEMS * 2);
  unsigned short* zerobuf = (unsigned short*)(ws + YELEMS * 2 + XELEMS * 2 + WELEMS * 2);
  float* psum = (float*)((char*)zerobuf + 64);
  float* pmax = psum + NIMG * 16;
  float* att  = pmax + NIMG * 16;

  hipMemsetAsync(zerobuf, 0, 64, stream);
  xpose_kernel<<<dim3(NIMG * 32), dim3(256), 0, stream>>>(data, xbf);
  wprep_kernel<<<dim3((WELEMS + 255) / 256), dim3(256), 0, stream>>>(conv_w, wbf);
  conv_wmma_kernel<<<dim3(NIMG * 16), dim3(256), 0, stream>>>(
      xbf, wbf, conv_b, zerobuf, yws, psum, pmax);
  ta_att_kernel<<<dim3(B_), dim3(64), 0, stream>>>(psum, pmax, ta_w1, ta_w2, att);
  lif_kernel<<<dim3(B_ * 64), dim3(256), 0, stream>>>(yws, att, out);
}
